// ScaleDotProductAttention_4544075399561
// MI455X (gfx1250) — matrix operand order repaired and checked by reading, not compiled, not run on hardware
//
#include <hip/hip_runtime.h>
#include <math.h>

// Flash-attention-style fused kernel for
//   score = beta*softmax(QK^T/sqrt(D) + adj); score[diag]=alpha; out = score @ V
// MI455X (gfx1250, wave32): f16 WMMA (v_wmma_f32_16x16x32_f16), f32 accumulate.
// adj (536 MB) is streamed exactly once -> ~23 us HBM floor at 23.3 TB/s.
// K/V pre-converted to f16 in d_ws; K/V tiles staged with
// global_load_async_to_lds_b128 (ASYNCcnt) into double-buffered LDS.

typedef __attribute__((ext_vector_type(16))) _Float16 v16h;
typedef __attribute__((ext_vector_type(8)))  float    v8f;
typedef __attribute__((ext_vector_type(4)))  _Float16 h4;
typedef __attribute__((ext_vector_type(4)))  float    f4;

namespace {
constexpr int HEADS = 8;
constexpr int NNODE = 4096;
constexpr int DDIM  = 64;
constexpr int BN    = 32;          // keys per inner iteration
// async-staged buffers: strides in halves; byte strides multiple of 16 and
// gcd(stride,256)=16 -> conflict-free 16-lane fragment reads
constexpr int KS_STRIDE = 72;      // 144 B per K row (64 halves + pad)
constexpr int VT_STRIDE = 40;      // 80 B per V^T row (32 halves + pad)
constexpr int PB_STRIDE = 36;      // P tile row stride (halves)
}

union V16U { v16h v; h4 q4[4]; _Float16 e[16]; };

__device__ __forceinline__ v8f wmma_f16(v16h a, v16h b, v8f c) {
  // (neg_a, A, neg_b, B, c_mod, C, reuse_a, reuse_b)
  return __builtin_amdgcn_wmma_f32_16x16x32_f16(false, a, false, b, (short)0, c,
                                                false, false);
}

__device__ __forceinline__ void async_cp16(unsigned lds_off,
                                           unsigned long long gaddr) {
  asm volatile("global_load_async_to_lds_b128 %0, %1, off"
               :: "v"(lds_off), "v"(gaddr) : "memory");
}

// ---------------- pre-pass: K -> f16 row-major, V -> f16 transposed ---------
__global__ __launch_bounds__(256)
void cvt_k_f16(const float* __restrict__ K, _Float16* __restrict__ KF) {
  const size_t i4 = (size_t)blockIdx.x * 256 + threadIdx.x;   // float4 index
  f4 x = *(const f4*)(K + i4 * 4);
  h4 hx; hx.x=(_Float16)x.x; hx.y=(_Float16)x.y; hx.z=(_Float16)x.z; hx.w=(_Float16)x.w;
  *(h4*)(KF + i4 * 4) = hx;
}

__global__ __launch_bounds__(256)
void cvt_v_tr(const float* __restrict__ V, _Float16* __restrict__ VT) {
  const size_t i4   = (size_t)blockIdx.x * 256 + threadIdx.x; // float4 index
  const size_t elem = i4 * 4;                 // index into [H][N][D]
  const int    d    = (int)(elem & (DDIM - 1));
  const size_t hn   = elem >> 6;              // h*N + n
  const int    n    = (int)(hn & (NNODE - 1));
  const int    h    = (int)(hn >> 12);
  f4 x = *(const f4*)(V + i4 * 4);
  _Float16* base = VT + ((size_t)h * DDIM + d) * NNODE + n;
  base[0 * NNODE] = (_Float16)x.x;
  base[1 * NNODE] = (_Float16)x.y;
  base[2 * NNODE] = (_Float16)x.z;
  base[3 * NNODE] = (_Float16)x.w;
}

// ---------------- main kernel: async-staged, double-buffered ----------------
__global__ __launch_bounds__(128)
void flash_adj_attn_async(const float* __restrict__ Q,
                          const _Float16* __restrict__ KF,  // [h][n][64] f16
                          const _Float16* __restrict__ VT,  // [h][64][n] f16
                          const float* __restrict__ V32,
                          const float* __restrict__ ADJ,
                          const float* __restrict__ ALPHA,
                          const float* __restrict__ BETA,
                          float* __restrict__ OUT)
{
  __shared__ _Float16 sK[2 * BN * KS_STRIDE];     // double-buffered [key][d]
  __shared__ _Float16 sV[2 * DDIM * VT_STRIDE];   // double-buffered [d][key]
  __shared__ _Float16 sP[4 * 16 * PB_STRIDE];     // per-wave P tile

  const int tid  = threadIdx.x;
  const int wave = tid >> 5;
  const int lane = tid & 31;
  const int lo16 = lane & 15;
  const int hi   = lane >> 4;

  const int blocksPerHead = NNODE / 64;
  const int h   = blockIdx.x / blocksPerHead;
  const int q0  = (blockIdx.x % blocksPerHead) * 64;
  const int qr0 = q0 + wave * 16;

  const float alphaH = ALPHA[h];
  const float betaH  = BETA[h];

  const float* Qh = Q   + (size_t)h * NNODE * DDIM;
  const float* Vh = V32 + (size_t)h * NNODE * DDIM;
  const float* Ah = ADJ + (size_t)h * NNODE * NNODE;
  float*       Oh = OUT + (size_t)h * NNODE * DDIM;
  const unsigned long long KFh = (unsigned long long)(KF + (size_t)h * NNODE * DDIM);
  const unsigned long long VTh = (unsigned long long)(VT + (size_t)h * DDIM * NNODE);

  _Float16* sPw = sP + wave * 16 * PB_STRIDE;
  const unsigned sKoff = (unsigned)(unsigned long long)(const void*)sK;
  const unsigned sVoff = (unsigned)(unsigned long long)(const void*)sV;

  // ---- stage one 32-key tile (K: 32x128B rows, V^T: 64x64B rows) ----------
  auto stage = [&](int j0, int buf) {
    const unsigned kb = sKoff + (unsigned)buf * (BN * KS_STRIDE * 2);
    const unsigned vb = sVoff + (unsigned)buf * (DDIM * VT_STRIDE * 2);
    #pragma unroll
    for (int i = 0; i < 2; ++i) {
      const int id  = (wave * 2 + i) * 32 + lane;   // 0..255 16B-chunk id
      // K tile: row = key (8 chunks/row)
      const int kr = id >> 3, kc = id & 7;
      async_cp16(kb + (unsigned)(kr * (KS_STRIDE * 2) + kc * 16),
                 KFh + (unsigned long long)(j0 + kr) * (DDIM * 2) + kc * 16);
      // V^T tile: row = d (4 chunks/row)
      const int vr = id >> 2, vc = id & 3;
      async_cp16(vb + (unsigned)(vr * (VT_STRIDE * 2) + vc * 16),
                 VTh + (unsigned long long)vr * (NNODE * 2) + j0 * 2 + vc * 16);
    }
  };

  // ---- Q strip as two 16x32 f16 A-fragments (scale folded in) -------------
  v16h aq[2];
  {
    const int row = qr0 + lo16;
    const int kb  = hi * 16;
    #pragma unroll
    for (int c = 0; c < 2; ++c) {
      const float* p0 = Qh + (size_t)row * DDIM + c * 32 + kb;
      f4 x0 = *(const f4*)(p0 + 0);
      f4 x1 = *(const f4*)(p0 + 4);
      f4 x2 = *(const f4*)(p0 + 8);
      f4 x3 = *(const f4*)(p0 + 12);
      V16U t;
      t.e[0]=(_Float16)(x0.x*0.125f); t.e[1]=(_Float16)(x0.y*0.125f);
      t.e[2]=(_Float16)(x0.z*0.125f); t.e[3]=(_Float16)(x0.w*0.125f);
      t.e[4]=(_Float16)(x1.x*0.125f); t.e[5]=(_Float16)(x1.y*0.125f);
      t.e[6]=(_Float16)(x1.z*0.125f); t.e[7]=(_Float16)(x1.w*0.125f);
      t.e[8]=(_Float16)(x2.x*0.125f); t.e[9]=(_Float16)(x2.y*0.125f);
      t.e[10]=(_Float16)(x2.z*0.125f); t.e[11]=(_Float16)(x2.w*0.125f);
      t.e[12]=(_Float16)(x3.x*0.125f); t.e[13]=(_Float16)(x3.y*0.125f);
      t.e[14]=(_Float16)(x3.z*0.125f); t.e[15]=(_Float16)(x3.w*0.125f);
      aq[c] = t.v;
    }
  }

  float mrow[8], lrow[8];
  v8f   oacc[4];
  #pragma unroll
  for (int r = 0; r < 8; ++r) { mrow[r] = -3.0e38f; lrow[r] = 0.f; }
  #pragma unroll
  for (int dc = 0; dc < 4; ++dc) oacc[dc] = (v8f){0,0,0,0,0,0,0,0};
  float dnn = 0.f;
  const int rdiag = lane & 7;

  // prologue: stage tile 0, wait, sync
  stage(0, 0);
  asm volatile("s_wait_asynccnt 0x0" ::: "memory");
  __syncthreads();

  int buf = 0;
  for (int j0 = 0; j0 < NNODE; j0 += BN) {
    // prefetch next tile into the other buffer (overlaps with compute below)
    if (j0 + BN < NNODE) stage(j0 + BN, buf ^ 1);

    const _Float16* sKc = sK + buf * (BN * KS_STRIDE);
    const _Float16* sVc = sV + buf * (DDIM * VT_STRIDE);

    // ---- S = Q * K^T (two 16x16 tiles, K-dim 64 -> 2 chunks) -------------
    v8f s0 = (v8f){0,0,0,0,0,0,0,0};
    v8f s1 = (v8f){0,0,0,0,0,0,0,0};
    #pragma unroll
    for (int c = 0; c < 2; ++c) {
      const int d0 = c * 32 + hi * 16;
      V16U b0, b1;
      #pragma unroll
      for (int t4 = 0; t4 < 4; ++t4) {
        b0.q4[t4] = *(const h4*)(&sKc[(0  + lo16) * KS_STRIDE + d0 + 4 * t4]);
        b1.q4[t4] = *(const h4*)(&sKc[(16 + lo16) * KS_STRIDE + d0 + 4 * t4]);
      }
      s0 = wmma_f16(aq[c], b0.v, s0);
      s1 = wmma_f16(aq[c], b1.v, s1);
    }

    // ---- bias + online softmax (C layout: row m=r+8*hi, col=lane%16) -----
    float p0[8], p1[8];
    #pragma unroll
    for (int r = 0; r < 8; ++r) {
      const int m = r + hi * 8;
      const float* ap = Ah + (size_t)(qr0 + m) * NNODE + j0 + lo16;
      p0[r] = s0[r] + ap[0];
      p1[r] = s1[r] + ap[16];
    }
    #pragma unroll
    for (int r = 0; r < 8; ++r) {
      const int m = r + hi * 8;
      float mx = fmaxf(p0[r], p1[r]);
      #pragma unroll
      for (int off = 1; off < 16; off <<= 1)
        mx = fmaxf(mx, __shfl_xor(mx, off, 32));
      const float mnew = fmaxf(mrow[r], mx);
      const float corr = __expf(mrow[r] - mnew);
      const float e0 = __expf(p0[r] - mnew);
      const float e1 = __expf(p1[r] - mnew);
      float rs = e0 + e1;
      #pragma unroll
      for (int off = 1; off < 16; off <<= 1)
        rs += __shfl_xor(rs, off, 32);
      lrow[r] = lrow[r] * corr + rs;
      mrow[r] = mnew;
      p0[r] = e0; p1[r] = e1;
      #pragma unroll
      for (int dc = 0; dc < 4; ++dc) oacc[dc][r] *= corr;
      if (r == rdiag) {
        dnn *= corr;
        if (j0 + lo16 == qr0 + m)       dnn += e0;
        if (j0 + 16 + lo16 == qr0 + m)  dnn += e1;
      }
    }

    // ---- P tile -> LDS (f16), O += P * V ---------------------------------
    #pragma unroll
    for (int r = 0; r < 8; ++r) {
      const int m = r + hi * 8;
      sPw[m * PB_STRIDE + lo16]      = (_Float16)p0[r];
      sPw[m * PB_STRIDE + 16 + lo16] = (_Float16)p1[r];
    }
    asm volatile("s_wait_dscnt 0" ::: "memory");  // same-wave LDS RAW

    V16U ap;
    {
      const int base = hi * 8;
      ap.q4[0] = *(const h4*)(&sPw[lo16 * PB_STRIDE + base + 0]);
      ap.q4[1] = *(const h4*)(&sPw[lo16 * PB_STRIDE + base + 4]);
      ap.q4[2] = *(const h4*)(&sPw[lo16 * PB_STRIDE + base + 16]);
      ap.q4[3] = *(const h4*)(&sPw[lo16 * PB_STRIDE + base + 20]);
    }
    #pragma unroll
    for (int dc = 0; dc < 4; ++dc) {
      const int dcol = dc * 16 + lo16;
      V16U bv;
      #pragma unroll
      for (int t4 = 0; t4 < 4; ++t4)
        bv.q4[t4] = *(const h4*)(&sVc[dcol * VT_STRIDE + hi * 8 + (t4 & 1) * 4 + (t4 >> 1) * 16]);
      oacc[dc] = wmma_f16(ap.v, bv.v, oacc[dc]);
    }

    // next tile must have arrived; everyone done with current buffers
    asm volatile("s_wait_asynccnt 0x0" ::: "memory");
    __syncthreads();
    buf ^= 1;
  }

  // ---- epilogue: normalize, beta/alpha, diagonal fix ----------------------
  float invl[8];
  #pragma unroll
  for (int r = 0; r < 8; ++r) invl[r] = 1.0f / lrow[r];
  const float coefLane = alphaH - betaH * dnn * invl[rdiag];

  #pragma unroll
  for (int r = 0; r < 8; ++r) {
    const int src  = (hi == 0) ? r : (24 + r);
    const float cf = __shfl(coefLane, src, 32);
    const int   R  = qr0 + r + hi * 8;
    const float* vr = Vh + (size_t)R * DDIM;
    float* orow     = Oh + (size_t)R * DDIM;
    #pragma unroll
    for (int dc = 0; dc < 4; ++dc) {
      const int d = dc * 16 + lo16;
      orow[d] = betaH * oacc[dc][r] * invl[r] + cf * vr[d];
    }
  }
}

// ---------------- fallback: self-contained (no workspace) ------------------
__global__ __launch_bounds__(128)
void flash_adj_attn(const float* __restrict__ Q,
                    const float* __restrict__ K,
                    const float* __restrict__ V,
                    const float* __restrict__ ADJ,
                    const float* __restrict__ ALPHA,
                    const float* __restrict__ BETA,
                    float* __restrict__ OUT)
{
  constexpr int KS2 = 72, VT2 = 36;
  __shared__ _Float16 sK[BN * KS2];
  __shared__ _Float16 sV[DDIM * VT2];
  __shared__ _Float16 sP[4 * 16 * PB_STRIDE];

  const int tid  = threadIdx.x;
  const int wave = tid >> 5;
  const int lane = tid & 31;
  const int lo16 = lane & 15;
  const int hi   = lane >> 4;

  const int blocksPerHead = NNODE / 64;
  const int h   = blockIdx.x / blocksPerHead;
  const int q0  = (blockIdx.x % blocksPerHead) * 64;
  const int qr0 = q0 + wave * 16;

  const float alphaH = ALPHA[h];
  const float betaH  = BETA[h];

  const size_t headOff = (size_t)h * NNODE * DDIM;
  const float* Qh = Q + headOff;
  const float* Kh = K + headOff;
  const float* Vh = V + headOff;
  const float* Ah = ADJ + (size_t)h * NNODE * NNODE;
  float*       Oh = OUT + headOff;
  _Float16* sPw = sP + wave * 16 * PB_STRIDE;

  v16h aq[2];
  {
    const int row = qr0 + lo16;
    const int kb  = hi * 16;
    #pragma unroll
    for (int c = 0; c < 2; ++c) {
      const float* p0 = Qh + (size_t)row * DDIM + c * 32 + kb;
      f4 x0 = *(const f4*)(p0 + 0);
      f4 x1 = *(const f4*)(p0 + 4);
      f4 x2 = *(const f4*)(p0 + 8);
      f4 x3 = *(const f4*)(p0 + 12);
      V16U t;
      t.e[0]=(_Float16)(x0.x*0.125f); t.e[1]=(_Float16)(x0.y*0.125f);
      t.e[2]=(_Float16)(x0.z*0.125f); t.e[3]=(_Float16)(x0.w*0.125f);
      t.e[4]=(_Float16)(x1.x*0.125f); t.e[5]=(_Float16)(x1.y*0.125f);
      t.e[6]=(_Float16)(x1.z*0.125f); t.e[7]=(_Float16)(x1.w*0.125f);
      t.e[8]=(_Float16)(x2.x*0.125f); t.e[9]=(_Float16)(x2.y*0.125f);
      t.e[10]=(_Float16)(x2.z*0.125f); t.e[11]=(_Float16)(x2.w*0.125f);
      t.e[12]=(_Float16)(x3.x*0.125f); t.e[13]=(_Float16)(x3.y*0.125f);
      t.e[14]=(_Float16)(x3.z*0.125f); t.e[15]=(_Float16)(x3.w*0.125f);
      aq[c] = t.v;
    }
  }

  float mrow[8], lrow[8];
  v8f   oacc[4];
  #pragma unroll
  for (int r = 0; r < 8; ++r) { mrow[r] = -3.0e38f; lrow[r] = 0.f; }
  #pragma unroll
  for (int dc = 0; dc < 4; ++dc) oacc[dc] = (v8f){0,0,0,0,0,0,0,0};
  float dnn = 0.f;
  const int rdiag = lane & 7;

  for (int j0 = 0; j0 < NNODE; j0 += BN) {
    __syncthreads();
    #pragma unroll
    for (int i = 0; i < 4; ++i) {
      const int idx = tid + i * 128;
      const int key = idx >> 4;
      const int dq  = (idx & 15) << 2;
      f4 x = *(const f4*)(Kh + (size_t)(j0 + key) * DDIM + dq);
      h4 hx; hx.x=(_Float16)x.x; hx.y=(_Float16)x.y; hx.z=(_Float16)x.z; hx.w=(_Float16)x.w;
      *(h4*)(&sK[key * KS2 + dq]) = hx;
    }
    #pragma unroll
    for (int i = 0; i < 4; ++i) {
      const int idx = tid + i * 128;
      const int key = idx >> 4;
      const int dq  = (idx & 15) << 2;
      f4 x = *(const f4*)(Vh + (size_t)(j0 + key) * DDIM + dq);
      sV[(dq + 0) * VT2 + key] = (_Float16)x.x;
      sV[(dq + 1) * VT2 + key] = (_Float16)x.y;
      sV[(dq + 2) * VT2 + key] = (_Float16)x.z;
      sV[(dq + 3) * VT2 + key] = (_Float16)x.w;
    }
    __syncthreads();

    v8f s0 = (v8f){0,0,0,0,0,0,0,0};
    v8f s1 = (v8f){0,0,0,0,0,0,0,0};
    #pragma unroll
    for (int c = 0; c < 2; ++c) {
      const int d0 = c * 32 + hi * 16;
      V16U b0, b1;
      #pragma unroll
      for (int t4 = 0; t4 < 4; ++t4) {
        b0.q4[t4] = *(const h4*)(&sK[(0  + lo16) * KS2 + d0 + 4 * t4]);
        b1.q4[t4] = *(const h4*)(&sK[(16 + lo16) * KS2 + d0 + 4 * t4]);
      }
      s0 = wmma_f16(aq[c], b0.v, s0);
      s1 = wmma_f16(aq[c], b1.v, s1);
    }

    float p0[8], p1[8];
    #pragma unroll
    for (int r = 0; r < 8; ++r) {
      const int m = r + hi * 8;
      const float* ap = Ah + (size_t)(qr0 + m) * NNODE + j0 + lo16;
      p0[r] = s0[r] + ap[0];
      p1[r] = s1[r] + ap[16];
    }
    #pragma unroll
    for (int r = 0; r < 8; ++r) {
      const int m = r + hi * 8;
      float mx = fmaxf(p0[r], p1[r]);
      #pragma unroll
      for (int off = 1; off < 16; off <<= 1)
        mx = fmaxf(mx, __shfl_xor(mx, off, 32));
      const float mnew = fmaxf(mrow[r], mx);
      const float corr = __expf(mrow[r] - mnew);
      const float e0 = __expf(p0[r] - mnew);
      const float e1 = __expf(p1[r] - mnew);
      float rs = e0 + e1;
      #pragma unroll
      for (int off = 1; off < 16; off <<= 1)
        rs += __shfl_xor(rs, off, 32);
      lrow[r] = lrow[r] * corr + rs;
      mrow[r] = mnew;
      p0[r] = e0; p1[r] = e1;
      #pragma unroll
      for (int dc = 0; dc < 4; ++dc) oacc[dc][r] *= corr;
      if (r == rdiag) {
        dnn *= corr;
        if (j0 + lo16 == qr0 + m)       dnn += e0;
        if (j0 + 16 + lo16 == qr0 + m)  dnn += e1;
      }
    }

    #pragma unroll
    for (int r = 0; r < 8; ++r) {
      const int m = r + hi * 8;
      sPw[m * PB_STRIDE + lo16]      = (_Float16)p0[r];
      sPw[m * PB_STRIDE + 16 + lo16] = (_Float16)p1[r];
    }
    asm volatile("s_wait_dscnt 0" ::: "memory");

    V16U ap;
    {
      const int base = hi * 8;
      ap.q4[0] = *(const h4*)(&sPw[lo16 * PB_STRIDE + base + 0]);
      ap.q4[1] = *(const h4*)(&sPw[lo16 * PB_STRIDE + base + 4]);
      ap.q4[2] = *(const h4*)(&sPw[lo16 * PB_STRIDE + base + 16]);
      ap.q4[3] = *(const h4*)(&sPw[lo16 * PB_STRIDE + base + 20]);
    }
    #pragma unroll
    for (int dc = 0; dc < 4; ++dc) {
      const int dcol = dc * 16 + lo16;
      V16U bv;
      #pragma unroll
      for (int t4 = 0; t4 < 4; ++t4)
        bv.q4[t4] = *(const h4*)(&sV[dcol * VT2 + hi * 8 + (t4 & 1) * 4 + (t4 >> 1) * 16]);
      oacc[dc] = wmma_f16(ap.v, bv.v, oacc[dc]);
    }
  }

  float invl[8];
  #pragma unroll
  for (int r = 0; r < 8; ++r) invl[r] = 1.0f / lrow[r];
  const float coefLane = alphaH - betaH * dnn * invl[rdiag];

  #pragma unroll
  for (int r = 0; r < 8; ++r) {
    const int src  = (hi == 0) ? r : (24 + r);
    const float cf = __shfl(coefLane, src, 32);
    const int   R  = qr0 + r + hi * 8;
    const float* vr = Vh + (size_t)R * DDIM;
    float* orow     = Oh + (size_t)R * DDIM;
    #pragma unroll
    for (int dc = 0; dc < 4; ++dc) {
      const int d = dc * 16 + lo16;
      orow[d] = betaH * oacc[dc][r] * invl[r] + cf * vr[d];
    }
  }
}

extern "C" void kernel_launch(void* const* d_in, const int* in_sizes, int n_in,
                              void* d_out, int out_size, void* d_ws, size_t ws_size,
                              hipStream_t stream) {
  const float* q     = (const float*)d_in[0];
  const float* k     = (const float*)d_in[1];
  const float* v     = (const float*)d_in[2];
  const float* adj   = (const float*)d_in[3];
  const float* alpha = (const float*)d_in[4];
  const float* beta  = (const float*)d_in[5];
  float* out = (float*)d_out;

  const size_t elems  = (size_t)HEADS * NNODE * DDIM;      // 2M
  const size_t needWS = 2 * elems * 2;                     // K f16 + V^T f16

  dim3 grid(HEADS * (NNODE / 64));   // 512 blocks, 4 waves each
  dim3 block(128);

  if (ws_size >= needWS) {
    _Float16* kf = (_Float16*)d_ws;
    _Float16* vt = kf + elems;
    const int cvtBlocks = (int)(elems / 4 / 256);          // 2048
    cvt_k_f16<<<cvtBlocks, 256, 0, stream>>>(k, kf);
    cvt_v_tr <<<cvtBlocks, 256, 0, stream>>>(v, vt);
    flash_adj_attn_async<<<grid, block, 0, stream>>>(q, kf, vt, v, adj,
                                                     alpha, beta, out);
  } else {
    flash_adj_attn<<<grid, block, 0, stream>>>(q, k, v, adj, alpha, beta, out);
  }
}
